// SynapseModel_29540785062023
// MI455X (gfx1250) — matrix operand order repaired and checked by reading, not compiled, not run on hardware
//
#include <hip/hip_runtime.h>

typedef __attribute__((ext_vector_type(16))) __bf16 v16bf;
typedef __attribute__((ext_vector_type(8)))  __bf16 v8bf;
typedef __attribute__((ext_vector_type(8)))  float  v8f;

union FragAB { v16bf v; v8bf h[2]; };

static __device__ inline __bf16 f2bf(float f) {
    union { float f; unsigned u; } a; a.f = f;
    unsigned r = a.u + 0x7FFFu + ((a.u >> 16) & 1u);   // round-to-nearest-even
    unsigned short hs = (unsigned short)(r >> 16);
    union { unsigned short s; __bf16 b; } o; o.s = hs;
    return o.b;
}

// ---------------------------------------------------------------------------
// Weight preprocessing
// ---------------------------------------------------------------------------
__global__ void absmean_partial(const float* __restrict__ W, long long n,
                                float* __restrict__ partial) {
    float s = 0.f;
    for (long long i = (long long)blockIdx.x * blockDim.x + threadIdx.x; i < n;
         i += (long long)gridDim.x * blockDim.x) {
        float w = fminf(fmaxf(W[i], -2.f), 2.f);
        s += fabsf(w);
    }
    __shared__ float sm[256];
    sm[threadIdx.x] = s; __syncthreads();
    for (int o = 128; o > 0; o >>= 1) {
        if (threadIdx.x < o) sm[threadIdx.x] += sm[threadIdx.x + o];
        __syncthreads();
    }
    if (threadIdx.x == 0) partial[blockIdx.x] = sm[0];
}

__global__ void reduce_final(const float* __restrict__ p, int n, float* __restrict__ out) {
    float s = 0.f;
    for (int i = threadIdx.x; i < n; i += 256) s += p[i];
    __shared__ float sm[256];
    sm[threadIdx.x] = s; __syncthreads();
    for (int o = 128; o > 0; o >>= 1) {
        if (threadIdx.x < o) sm[threadIdx.x] += sm[threadIdx.x + o];
        __syncthreads();
    }
    if (threadIdx.x == 0) out[0] = sm[0];
}

__global__ void quantize_to_bf16(const float* __restrict__ W, __bf16* __restrict__ Q,
                                 long long n, const float* __restrict__ sum) {
    float gamma = fmaxf(sum[0] / (float)n, 1e-4f);
    float inv = 1.f / gamma;
    for (long long i = (long long)blockIdx.x * blockDim.x + threadIdx.x; i < n;
         i += (long long)gridDim.x * blockDim.x) {
        float w = fminf(fmaxf(W[i], -2.f), 2.f);
        float q = fminf(fmaxf(rintf(w * inv), -1.f), 1.f) * gamma;
        Q[i] = f2bf(q);
    }
}

__global__ void convert_to_bf16(const float* __restrict__ W, __bf16* __restrict__ Q,
                                long long n) {
    for (long long i = (long long)blockIdx.x * blockDim.x + threadIdx.x; i < n;
         i += (long long)gridDim.x * blockDim.x)
        Q[i] = f2bf(W[i]);
}

// attention_out fp32 [N,H] -> combined bf16 [N,2H] columns H..2H-1
__global__ void concat_attn(const float* __restrict__ X, __bf16* __restrict__ C,
                            int N, int H) {
    long long n = (long long)N * H;
    for (long long i = (long long)blockIdx.x * blockDim.x + threadIdx.x; i < n;
         i += (long long)gridDim.x * blockDim.x) {
        long long row = i / H; int col = (int)(i - row * H);
        C[row * (2LL * H) + H + col] = f2bf(X[i]);
    }
}

// ---------------------------------------------------------------------------
// LayerNorm (+optional exact GELU) over H=2048, one block per row, bf16 out
// ---------------------------------------------------------------------------
__global__ __launch_bounds__(256)
void ln_act(const float* __restrict__ X, const float* __restrict__ g,
            const float* __restrict__ b, __bf16* __restrict__ Y,
            int outStride, int outOff, int doGelu) {
    const int H = 2048;
    const int row = blockIdx.x;
    const float* x = X + (long long)row * H;
    float vals[8];
    float s = 0.f, ss = 0.f;
#pragma unroll
    for (int i = 0; i < 8; ++i) {
        float v = x[i * 256 + threadIdx.x];
        vals[i] = v; s += v; ss += v * v;
    }
    __shared__ float sm[256], sv[256];
    sm[threadIdx.x] = s; sv[threadIdx.x] = ss; __syncthreads();
    for (int o = 128; o > 0; o >>= 1) {
        if (threadIdx.x < o) { sm[threadIdx.x] += sm[threadIdx.x + o];
                               sv[threadIdx.x] += sv[threadIdx.x + o]; }
        __syncthreads();
    }
    float mean = sm[0] * (1.f / H);
    float var  = sv[0] * (1.f / H) - mean * mean;
    float rstd = rsqrtf(var + 1e-5f);
    __bf16* y = Y + (long long)row * outStride + outOff;
#pragma unroll
    for (int i = 0; i < 8; ++i) {
        int j = i * 256 + threadIdx.x;
        float v = (vals[i] - mean) * rstd * g[j] + b[j];
        if (doGelu) v = 0.5f * v * (1.f + erff(v * 0.70710678118654752f));
        y[j] = f2bf(v);
    }
}

// ---------------------------------------------------------------------------
// bf16 WMMA GEMM:  C[M,N] = A[M,K] * B[N,K]^T + bias[N]   (fp32 accumulate)
// Block = 128 threads (4 waves, 2x2), block tile 128x128, wave tile 64x64,
// K-step 32, double-buffered LDS, v_wmma_f32_16x16x32_bf16.
// ---------------------------------------------------------------------------
#define LDSS 40   // padded row stride (elems) -> conflict-free 16B fragment reads

template <bool AFP32>
__global__ __launch_bounds__(128)
void gemm_bf16_wmma(const void* __restrict__ Av, const __bf16* __restrict__ B,
                    const float* __restrict__ bias, float* __restrict__ C,
                    int M, int N, int K) {
    __shared__ __attribute__((aligned(16))) __bf16 lA[2][128 * LDSS];
    __shared__ __attribute__((aligned(16))) __bf16 lB[2][128 * LDSS];

    const int tid  = threadIdx.x;
    const int lane = tid & 31;
    const int wave = tid >> 5;
    const int waveRow = (wave >> 1) * 64;
    const int waveCol = (wave & 1) * 64;
    const int blockRow = blockIdx.y * 128;
    const int blockCol = blockIdx.x * 128;
    const int half = lane >> 4;
    const int r16  = lane & 15;

    const float*  Af = (const float*)Av;
    const __bf16* Ab = (const __bf16*)Av;

    v8f acc[4][4];
#pragma unroll
    for (int i = 0; i < 4; ++i)
#pragma unroll
        for (int j = 0; j < 4; ++j) acc[i][j] = v8f{};

    float4 raf[8];   // fp32-A staging
    uint4  rab[4];   // bf16-A staging
    uint4  rbb[4];   // B staging

    auto ldg = [&](int k0) {
        if (AFP32) {
#pragma unroll
            for (int i = 0; i < 8; ++i) {
                int c = i * 128 + tid; int row = c >> 3; int kc = c & 7;
                raf[i] = *(const float4*)(Af + (size_t)(blockRow + row) * K + k0 + kc * 4);
            }
        } else {
#pragma unroll
            for (int i = 0; i < 4; ++i) {
                int c = i * 128 + tid; int row = c >> 2; int kc = c & 3;
                rab[i] = *(const uint4*)(Ab + (size_t)(blockRow + row) * K + k0 + kc * 8);
            }
        }
#pragma unroll
        for (int i = 0; i < 4; ++i) {
            int c = i * 128 + tid; int row = c >> 2; int kc = c & 3;
            rbb[i] = *(const uint4*)(B + (size_t)(blockCol + row) * K + k0 + kc * 8);
        }
    };

    auto sts = [&](int buf) {
        if (AFP32) {
#pragma unroll
            for (int i = 0; i < 8; ++i) {
                int c = i * 128 + tid; int row = c >> 3; int kc = c & 7;
                union { __bf16 b[4]; uint2 u; } p;
                p.b[0] = f2bf(raf[i].x); p.b[1] = f2bf(raf[i].y);
                p.b[2] = f2bf(raf[i].z); p.b[3] = f2bf(raf[i].w);
                *(uint2*)(&lA[buf][row * LDSS + kc * 4]) = p.u;
            }
        } else {
#pragma unroll
            for (int i = 0; i < 4; ++i) {
                int c = i * 128 + tid; int row = c >> 2; int kc = c & 3;
                *(uint4*)(&lA[buf][row * LDSS + kc * 8]) = rab[i];
            }
        }
#pragma unroll
        for (int i = 0; i < 4; ++i) {
            int c = i * 128 + tid; int row = c >> 2; int kc = c & 3;
            *(uint4*)(&lB[buf][row * LDSS + kc * 8]) = rbb[i];
        }
    };

    auto compute = [&](int buf) {
        FragAB a[4], bfr[4];
#pragma unroll
        for (int mi = 0; mi < 4; ++mi) {
            // A 16x32 bf16 layout: lane<16 -> row=lane, K{0..7,16..23}; lane>=16 -> K{8..15,24..31}
            int base = (waveRow + mi * 16 + r16) * LDSS + half * 8;
            a[mi].h[0] = *(const v8bf*)(&lA[buf][base]);
            a[mi].h[1] = *(const v8bf*)(&lA[buf][base + 16]);
        }
#pragma unroll
        for (int nj = 0; nj < 4; ++nj) {
            // B 32x16 bf16 layout: lane holds col n=lane%16, lane<16 -> K{0..7,16..23}; lane>=16 -> K{8..15,24..31}
            int base = (waveCol + nj * 16 + r16) * LDSS + half * 8;
            bfr[nj].h[0] = *(const v8bf*)(&lB[buf][base]);
            bfr[nj].h[1] = *(const v8bf*)(&lB[buf][base + 16]);
        }
#pragma unroll
        for (int mi = 0; mi < 4; ++mi)
#pragma unroll
            for (int nj = 0; nj < 4; ++nj)
                acc[mi][nj] = __builtin_amdgcn_wmma_f32_16x16x32_bf16(
                    false, a[mi].v, false, bfr[nj].v, (short)0, acc[mi][nj],
                    false, false);
    };

    const int nk = K >> 5;
    ldg(0);
    sts(0);
    __syncthreads();
    for (int kt = 0; kt < nk; ++kt) {
        if (kt + 1 < nk) ldg((kt + 1) << 5);
        compute(kt & 1);
        if (kt + 1 < nk) sts((kt + 1) & 1);
        __syncthreads();
    }

    // Epilogue: C/D layout -> VGPR e: lanes 0-15 M=e, lanes 16-31 M=e+8; N=lane%16
#pragma unroll
    for (int nj = 0; nj < 4; ++nj) {
        int col = blockCol + waveCol + nj * 16 + r16;
        float bia = bias[col];
#pragma unroll
        for (int mi = 0; mi < 4; ++mi) {
            int rowb = blockRow + waveRow + mi * 16 + half * 8;
#pragma unroll
            for (int e = 0; e < 8; ++e)
                C[(size_t)(rowb + e) * N + col] = acc[mi][nj][e] + bia;
        }
    }
}

// ---------------------------------------------------------------------------
// Launch
// ---------------------------------------------------------------------------
extern "C" void kernel_launch(void* const* d_in, const int* in_sizes, int n_in,
                              void* d_out, int out_size, void* d_ws, size_t ws_size,
                              hipStream_t stream) {
    const int N = 16384, D = 4096, H = 2048;

    const float* z     = (const float*)d_in[0];
    const float* attn  = (const float*)d_in[1];
    const float* Wz    = (const float*)d_in[2];
    const float* bz    = (const float*)d_in[3];
    const float* gz    = (const float*)d_in[4];
    const float* bzln  = (const float*)d_in[5];
    const float* W1    = (const float*)d_in[6];
    const float* b1    = (const float*)d_in[7];
    const float* g1    = (const float*)d_in[8];
    const float* b1ln  = (const float*)d_in[9];
    const float* W2    = (const float*)d_in[10];
    const float* b2    = (const float*)d_in[11];
    const float* g2    = (const float*)d_in[12];
    const float* b2ln  = (const float*)d_in[13];
    const float* Wo    = (const float*)d_in[14];
    const float* bo    = (const float*)d_in[15];

    char* w = (char*)d_ws;
    __bf16* WZB  = (__bf16*)(w);                          // 16 MB
    __bf16* W1Q  = (__bf16*)(w + ((size_t)16 << 20));     // 16 MB
    __bf16* W2Q  = (__bf16*)(w + ((size_t)32 << 20));     //  8 MB
    __bf16* WOB  = (__bf16*)(w + ((size_t)40 << 20));     // 16 MB
    float*  RED  = (float*) (w + ((size_t)56 << 20));     // partials + gammas
    float*  P1   = RED;                                   // 2048 partials
    float*  P2   = RED + 2048;                            // 2048 partials
    float*  GAM  = RED + 4096;                            // 2 sums
    float*  TMP  = (float*) (w + ((size_t)64  << 20));    // 128 MB fp32 [N,2048]
    __bf16* COMB = (__bf16*)(w + ((size_t)192 << 20));    // 128 MB bf16 [N,4096]
    __bf16* HB   = (__bf16*)(w + ((size_t)320 << 20));    //  64 MB bf16 [N,2048]

    const long long nW1 = (long long)H * 2 * H;
    const long long nW2 = (long long)H * H;

    // BitNet gamma (deterministic 2-stage reduction) + weight conversion
    absmean_partial<<<2048, 256, 0, stream>>>(W1, nW1, P1);
    absmean_partial<<<2048, 256, 0, stream>>>(W2, nW2, P2);
    reduce_final<<<1, 256, 0, stream>>>(P1, 2048, GAM + 0);
    reduce_final<<<1, 256, 0, stream>>>(P2, 2048, GAM + 1);
    convert_to_bf16<<<2048, 256, 0, stream>>>(Wz, WZB, (long long)H * D);
    quantize_to_bf16<<<2048, 256, 0, stream>>>(W1, W1Q, nW1, GAM + 0);
    quantize_to_bf16<<<2048, 256, 0, stream>>>(W2, W2Q, nW2, GAM + 1);
    convert_to_bf16<<<2048, 256, 0, stream>>>(Wo, WOB, (long long)D * H);

    dim3 blk(128);

    // Layer 0: z @ Wz^T + bz -> LN -> combined[:, :H]
    gemm_bf16_wmma<true><<<dim3(H / 128, N / 128), blk, 0, stream>>>(
        z, WZB, bz, TMP, N, H, D);
    ln_act<<<N, 256, 0, stream>>>(TMP, gz, bzln, COMB, 2 * H, 0, 0);
    concat_attn<<<2048, 256, 0, stream>>>(attn, COMB, N, H);

    // Layer 1: combined @ W1q^T + b1 -> LN -> GELU -> h
    gemm_bf16_wmma<false><<<dim3(H / 128, N / 128), blk, 0, stream>>>(
        COMB, W1Q, b1, TMP, N, H, 2 * H);
    ln_act<<<N, 256, 0, stream>>>(TMP, g1, b1ln, HB, H, 0, 1);

    // Layer 2: h @ W2q^T + b2 -> LN -> GELU -> h
    gemm_bf16_wmma<false><<<dim3(H / 128, N / 128), blk, 0, stream>>>(
        HB, W2Q, b2, TMP, N, H, H);
    ln_act<<<N, 256, 0, stream>>>(TMP, g2, b2ln, HB, H, 0, 1);

    // Output: h @ Wo^T + bo -> d_out (fp32)
    gemm_bf16_wmma<false><<<dim3(D / 128, N / 128), blk, 0, stream>>>(
        HB, WOB, bo, (float*)d_out, N, D, H);
}
